// TADCModule_64561948394028
// MI455X (gfx1250) — matrix operand order repaired and checked by reading, not compiled, not run on hardware
//
#include <hip/hip_runtime.h>

// MI455X (gfx1250) wave32 / WMMA implementation.
// Pipeline (all on `stream`):
//   k_accum   : acc = sum_t |raw_diff|, per-block partial sums for mean/var
//   k_stats   : deterministic tree-reduce -> scal[0]=1/(std+1e-6), scal[1]=-mean*scal0
//   k_conv1   : motion-normalize + collapsed (1-in-ch) 3x3 conv + BN + ReLU -> h1 (f16)
//   k_convwmma<DIL>: dilated 3x3 16->16 conv as implicit GEMM via v_wmma_f32_16x16x32_f16
//               (K = 16ch*9taps = 144 padded to 160 -> 5 WMMA per 16-out x 16-pixel tile)
//   k_final   : 1x1 conv (3 WMMAs) + sigmoid kernel + fused unfold weighted sum -> out

typedef __attribute__((ext_vector_type(16))) _Float16 v16h;
typedef __attribute__((ext_vector_type(8)))  float    v8f;

#define NB    8
#define NT    5
#define NH    512
#define NW    512
#define MID   16
#define PLANE (NH * NW)        // 262144
#define NPIX  (NB * PLANE)     // 2097152
#define KTOT  144              // 16 in-ch * 9 taps
#define BN_EPS 1e-5f

// ---------------------------------------------------------------- reductions
__global__ void k_accum(const float* __restrict__ diff, float* __restrict__ acc,
                        float* __restrict__ part, float* __restrict__ partsq) {
  int tid = threadIdx.x;
  int idx = blockIdx.x * 256 + tid;              // one pixel (b,h,w)
  int b   = idx >> 18;
  int hw  = idx & (PLANE - 1);
  const float* p = diff + (size_t)b * 4 * PLANE + hw;
  float s = fabsf(p[0]) + fabsf(p[PLANE]) + fabsf(p[2 * PLANE]) + fabsf(p[3 * PLANE]);
  acc[idx] = s;
  __shared__ float sh[256];
  __shared__ float sh2[256];
  sh[tid] = s; sh2[tid] = s * s;
  __syncthreads();
  for (int off = 128; off > 0; off >>= 1) {
    if (tid < off) { sh[tid] += sh[tid + off]; sh2[tid] += sh2[tid + off]; }
    __syncthreads();
  }
  if (tid == 0) { part[blockIdx.x] = sh[0]; partsq[blockIdx.x] = sh2[0]; }
}

__global__ void k_stats(const float* __restrict__ part, const float* __restrict__ partsq,
                        float* __restrict__ scal) {
  int tid = threadIdx.x;
  float s = 0.f, s2 = 0.f;
  for (int i = tid; i < 8192; i += 256) { s += part[i]; s2 += partsq[i]; }
  __shared__ float sh[256];
  __shared__ float sh2[256];
  sh[tid] = s; sh2[tid] = s2;
  __syncthreads();
  for (int off = 128; off > 0; off >>= 1) {
    if (tid < off) { sh[tid] += sh[tid + off]; sh2[tid] += sh2[tid + off]; }
    __syncthreads();
  }
  if (tid == 0) {
    const float n = (float)NPIX;
    float mean = sh[0] / n;
    float var  = (sh2[0] - n * mean * mean) / (n - 1.f);   // unbiased
    float inv  = 1.f / (sqrtf(var) + 1e-6f);
    scal[0] = inv;
    scal[1] = -mean * inv;
  }
}

// ------------------------------------------------------- conv1 (collapsed)
// motion_seq has T identical channels -> conv1 == 1-in-channel conv with
// weights summed over the input-channel axis.
__global__ void k_conv1(const float* __restrict__ acc, const float* __restrict__ scal,
                        const float* __restrict__ w1, const float* __restrict__ g1,
                        const float* __restrict__ b1, const float* __restrict__ rm1,
                        const float* __restrict__ rv1, _Float16* __restrict__ hout) {
  __shared__ float s_ws[16 * 9];
  __shared__ float s_bns[16];
  __shared__ float s_bno[16];
  int tid = threadIdx.x;
  if (tid < 144) {                                // channel-summed 3x3 weights
    int o = tid / 9, tap = tid % 9;
    float s = 0.f;
#pragma unroll
    for (int i = 0; i < NT; ++i) s += w1[(o * NT + i) * 9 + tap];
    s_ws[tid] = s;
  }
  if (tid < 16) {
    float inv = g1[tid] * rsqrtf(rv1[tid] + BN_EPS);
    s_bns[tid] = inv;
    s_bno[tid] = b1[tid] - rm1[tid] * inv;
  }
  __syncthreads();

  int idx = blockIdx.x * 256 + tid;
  int b = idx >> 18, rem = idx & (PLANE - 1);
  int y = rem >> 9, x = rem & (NW - 1);
  float s0 = scal[0], s1 = scal[1];

  float nb[9];
#pragma unroll
  for (int tap = 0; tap < 9; ++tap) {
    int gy = y + tap / 3 - 1, gx = x + tap % 3 - 1;
    bool in = (gy >= 0) & (gy < NH) & (gx >= 0) & (gx < NW);
    // padding applied to motion (value 0), not to acc
    nb[tap] = in ? (acc[(b << 18) + (gy << 9) + gx] * s0 + s1) : 0.f;
  }
#pragma unroll
  for (int o = 0; o < MID; ++o) {
    float v = 0.f;
#pragma unroll
    for (int tap = 0; tap < 9; ++tap) v += nb[tap] * s_ws[o * 9 + tap];
    v = fmaxf(v * s_bns[o] + s_bno[o], 0.f);
    hout[((b * MID + o) << 18) + (y << 9) + x] = (_Float16)v;
  }
}

// ------------------------------------------ dilated 3x3 conv via WMMA f16
// Workgroup = 8 waves. Tile = 8 rows x 16 cols, all 16 out channels.
// Per wave: D[16 out][16 px] += A[16 out][160 K] x B[160 K][16 px], 5 WMMAs.
// DIL is a template parameter so tile geometry constant-folds into immediate
// LDS offsets instead of runtime v_mad_u32 chains.
template <int DIL>
__global__ void k_convwmma(const _Float16* __restrict__ hin, const float* __restrict__ w,
                           const float* __restrict__ g, const float* __restrict__ bb,
                           const float* __restrict__ rm, const float* __restrict__ rv,
                           _Float16* __restrict__ hout) {
  constexpr int ROWS = 8 + 2 * DIL;
  constexpr int COLS = 16 + 2 * DIL;
  __shared__ _Float16 lds_in[MID * ROWS * COLS];  // [c][row][col], zero-padded halo
  __shared__ _Float16 lds_w[MID * KTOT];          // [o][K], K = c*9+tap (w already flat this way)
  __shared__ float s_bns[16];
  __shared__ float s_bno[16];

  const int tid   = threadIdx.x;
  const int bimg  = blockIdx.z;
  const int x0    = blockIdx.x * 16;
  const int ytile = blockIdx.y * 8;

  for (int i = tid; i < MID * KTOT; i += 256) lds_w[i] = (_Float16)w[i];
  if (tid < 16) {
    float inv = g[tid] * rsqrtf(rv[tid] + BN_EPS);
    s_bns[tid] = inv;
    s_bno[tid] = bb[tid] - rm[tid] * inv;
  }
  // stage zero-padded input halo tile: 16 ch x ROWS x COLS
  constexpr int total = MID * ROWS * COLS;
  for (int i = tid; i < total; i += 256) {
    int ci = i / (ROWS * COLS);
    int r  = i % (ROWS * COLS);
    int ry = r / COLS, rx = r % COLS;
    int gy = ytile - DIL + ry, gx = x0 - DIL + rx;
    _Float16 v = (_Float16)0;
    if (gy >= 0 && gy < NH && gx >= 0 && gx < NW)
      v = hin[((bimg * MID + ci) << 18) + (gy << 9) + gx];
    lds_in[(ci * ROWS + ry) * COLS + rx] = v;
  }
  __syncthreads();

  const int wave = tid >> 5, lane = tid & 31;
  const int m = lane & 15, hi = lane >> 4;
  const int y = ytile + wave;

  // A fragments: lane -> M=m; reg/half/lane-hi -> K (ISA 16-bit A 16x32 layout)
  v16h a[5];
#pragma unroll
  for (int q = 0; q < 5; ++q) {
#pragma unroll
    for (int i = 0; i < 16; ++i) {
      int K = q * 32 + hi * 8 + ((i < 8) ? i : (8 + i));   // i<8: K=hi*8+i ; i>=8: K=16+hi*8+(i-8)
      a[q][i] = (K < KTOT) ? lds_w[m * KTOT + K] : (_Float16)0;
    }
  }

  // Per-chunk B-fragment LDS base addresses (k = q*32+lane -> (ch, tap))
  int bbase[5];
  bool bvalid[5];
#pragma unroll
  for (int q = 0; q < 5; ++q) {
    int k = q * 32 + lane;
    bvalid[q] = (k < KTOT);
    int kk = bvalid[q] ? k : 0;
    int ci = kk / 9, tap = kk % 9;
    bbase[q] = (ci * ROWS + (wave + (tap / 3) * DIL)) * COLS + (tap % 3) * DIL;
  }

  v8f c = {};
#pragma unroll
  for (int q = 0; q < 5; ++q) {
    // B fragment: lane -> N (pixel) ; reg/lane-hi -> K (same order as A)
    v16h bf;
#pragma unroll
    for (int i = 0; i < 16; ++i) {
      int kb = q * 32 + hi * 8 + ((i < 8) ? i : (8 + i));
      int ci = kb / 9, tap = kb % 9;
      bf[i] = (kb < KTOT) ? lds_in[(ci * ROWS + (wave + (tap / 3) * DIL)) * COLS + (tap % 3) * DIL + m] : (_Float16)0;
    }
    c = __builtin_amdgcn_wmma_f32_16x16x32_f16(false, a[q], false, bf, (short)0, c,
                                               false, false);
  }

  // BN + ReLU fused on accumulator; C layout: reg r -> M = hi*8+r, lane -> N
#pragma unroll
  for (int r = 0; r < 8; ++r) {
    int mo = hi * 8 + r;
    float v = fmaxf(c[r] * s_bns[mo] + s_bno[mo], 0.f);
    hout[((bimg * MID + mo) << 18) + (y << 9) + x0 + (lane & 15)] = (_Float16)v;
  }
}

// ------------------- 1x1 conv (45 out) + sigmoid + fused unfold weighted sum
__global__ void k_final(const _Float16* __restrict__ h3, const float* __restrict__ w4,
                        const float* __restrict__ biasp, const float* __restrict__ x_al,
                        float* __restrict__ out) {
  __shared__ float lds_k[128][48];   // [tile pixel][kernel channel]

  const int tid   = threadIdx.x;
  const int bimg  = blockIdx.z;
  const int x0    = blockIdx.x * 16;
  const int ytile = blockIdx.y * 8;
  const int wave  = tid >> 5, lane = tid & 31;
  const int m = lane & 15, hi = lane >> 4;
  const int y = ytile + wave;
  const float bias = biasp[0];

  // warm L2/L0 for the x_aligned tile we will read after the WMMAs
  for (int task = tid; task < 640; task += 256) {
    int t = task >> 7, p = task & 127;
    int py = ytile + (p >> 4), px = x0 + (p & 15);
    __builtin_prefetch(x_al + (((bimg * NT + t)) << 18) + (py << 9) + px, 0, 0);
  }

  // B fragment: lane -> pixel; reg/lane-hi -> K (in channel, 16 real + 16 zero pad)
  v16h bf;
#pragma unroll
  for (int i = 0; i < 16; ++i) {
    int K = hi * 8 + ((i < 8) ? i : (8 + i));
    bf[i] = (K < MID) ? h3[((bimg * MID + K) << 18) + (y << 9) + x0 + m] : (_Float16)0;
  }

#pragma unroll
  for (int t = 0; t < 3; ++t) {                 // out channels [16t, 16t+16)
    v16h af;
#pragma unroll
    for (int i = 0; i < 16; ++i) {
      int K  = hi * 8 + ((i < 8) ? i : (8 + i));
      int oc = t * 16 + m;
      af[i] = (K < MID && oc < NT * 9) ? (_Float16)w4[oc * MID + K] : (_Float16)0;
    }
    v8f cz = {};
    v8f c = __builtin_amdgcn_wmma_f32_16x16x32_f16(false, af, false, bf, (short)0, cz,
                                                   false, false);
#pragma unroll
    for (int r = 0; r < 8; ++r) {
      int ch = t * 16 + hi * 8 + r;
      if (ch < NT * 9) {
        float lg = c[r] + bias;
        // sigmoid with hardware reciprocal (v_rcp_f32) instead of IEEE divide
        float sg = __builtin_amdgcn_rcpf(1.f + __expf(-lg));
        lds_k[wave * 16 + (lane & 15)][ch] = (sg + 0.01f) * 10.f;
      }
    }
  }
  __syncthreads();

  // fused unfold + weighted sum: 640 (pixel, t) tasks over 256 threads
  for (int task = tid; task < 640; task += 256) {
    int t = task >> 7, p = task & 127;
    int py = ytile + (p >> 4), px = x0 + (p & 15);
    const float* xb = x_al + ((bimg * NT + t) << 18);
    float acc = 0.f;
#pragma unroll
    for (int tap = 0; tap < 9; ++tap) {
      int gy = py + tap / 3 - 1, gx = px + tap % 3 - 1;
      bool in = (gy >= 0) & (gy < NH) & (gx >= 0) & (gx < NW);
      float xv = in ? xb[(gy << 9) + gx] : 0.f;
      acc += lds_k[p][t * 9 + tap] * xv;
    }
    out[((bimg * NT + t) << 18) + (py << 9) + px] = acc;
  }
}

// ------------------------------------------------------------------- launch
extern "C" void kernel_launch(void* const* d_in, const int* in_sizes, int n_in,
                              void* d_out, int out_size, void* d_ws, size_t ws_size,
                              hipStream_t stream) {
  const float* x_al = (const float*)d_in[0];
  const float* diff = (const float*)d_in[1];
  const float* w1   = (const float*)d_in[2];
  const float* g1   = (const float*)d_in[3];
  const float* b1   = (const float*)d_in[4];
  const float* rm1  = (const float*)d_in[5];
  const float* rv1  = (const float*)d_in[6];
  const float* w2   = (const float*)d_in[7];
  const float* g2   = (const float*)d_in[8];
  const float* b2   = (const float*)d_in[9];
  const float* rm2  = (const float*)d_in[10];
  const float* rv2  = (const float*)d_in[11];
  const float* w3   = (const float*)d_in[12];
  const float* g3   = (const float*)d_in[13];
  const float* b3   = (const float*)d_in[14];
  const float* rm3  = (const float*)d_in[15];
  const float* rv3  = (const float*)d_in[16];
  const float* w4   = (const float*)d_in[17];
  const float* bias = (const float*)d_in[18];

  char* ws = (char*)d_ws;
  float*    scal   = (float*)(ws + 0);
  float*    part   = (float*)(ws + 1024);
  float*    partsq = (float*)(ws + 1024 + 8192 * 4);
  float*    acc    = (float*)(ws + 131072);
  _Float16* hA     = (_Float16*)(ws + 131072 + (size_t)NPIX * 4);
  _Float16* hB     = (_Float16*)(ws + 131072 + (size_t)NPIX * 4 + (size_t)NPIX * MID * 2);

  k_accum<<<NPIX / 256, 256, 0, stream>>>(diff, acc, part, partsq);
  k_stats<<<1, 256, 0, stream>>>(part, partsq, scal);
  k_conv1<<<NPIX / 256, 256, 0, stream>>>(acc, scal, w1, g1, b1, rm1, rv1, hA);
  dim3 gridT(NW / 16, NH / 8, NB);
  k_convwmma<2><<<gridT, 256, 0, stream>>>(hA, w2, g2, b2, rm2, rv2, hB);
  k_convwmma<4><<<gridT, 256, 0, stream>>>(hB, w3, g3, b3, rm3, rv3, hA);
  k_final<<<gridT, 256, 0, stream>>>(hA, w4, bias, x_al, (float*)d_out);
}
